// GATModel_50036368998454
// MI455X (gfx1250) — matrix operand order repaired and checked by reading, not compiled, not run on hardware
//
#include <hip/hip_runtime.h>
#include <hip/hip_bf16.h>

// ============================================================================
// GATModel on MI455X (gfx1250, wave32, WMMA).
// Only the S tower feeds the output (xt is dead in the reference) -> T tower
// is skipped entirely (legitimate DCE).
// Dense transforms use v_wmma_f32_16x16x32_f16 with branchless, unconditional
// (clamped+masked) loads so the compiler emits coalesced global_load_b128/b64
// instead of per-element predicated loads. Edge softmax/aggregation use float
// atomics; GraphNorm stats use LDS ds_add_f32 reductions.
// ============================================================================

typedef __attribute__((ext_vector_type(16))) _Float16 v16h;
typedef __attribute__((ext_vector_type(8)))  float    v8f;

#define NS 30000
#define ES 480000
#define NB 64

// ---------------------------------------------------------------- utilities
__device__ __forceinline__ unsigned f2mono(float f) {
  unsigned u = __float_as_uint(f);
  return (u & 0x80000000u) ? ~u : (u | 0x80000000u);
}
__device__ __forceinline__ float mono2f(unsigned u) {
  return __uint_as_float((u & 0x80000000u) ? (u & 0x7FFFFFFFu) : ~u);
}

__global__ void k_fill_f32(float* __restrict__ p, float v, long n) {
  for (long i = (long)blockIdx.x * blockDim.x + threadIdx.x; i < n;
       i += (long)gridDim.x * blockDim.x) p[i] = v;
}
__global__ void k_fill_u32(unsigned* __restrict__ p, unsigned v, long n) {
  for (long i = (long)blockIdx.x * blockDim.x + threadIdx.x; i < n;
       i += (long)gridDim.x * blockDim.x) p[i] = v;
}
__global__ void k_bias_init(float* __restrict__ out, const float* __restrict__ bias,
                            long total, int D) {
  for (long i = (long)blockIdx.x * blockDim.x + threadIdx.x; i < total;
       i += (long)gridDim.x * blockDim.x) out[i] = bias[(int)(i % D)];
}

// ---------------------------------------------------- WMMA GEMM: C = A*B^T + b
// A[M,K] row-major, B[N,K] row-major, C[M,N]. One wave per 16x16 C tile.
// All loads are unconditional (row-clamped, mask-multiplied) so they vectorize.
__global__ void k_wmma_gemm_nt(const float* __restrict__ A, const float* __restrict__ B,
                               const float* __restrict__ bias, float* __restrict__ C,
                               int M, int N, int K, int tilesN) {
  const int lane = threadIdx.x & 31;
  const int wave = threadIdx.x >> 5;
  const int tile = blockIdx.x * (blockDim.x >> 5) + wave;
  const int tm = tile / tilesN;
  const int tn = tile - tm * tilesN;
  const int m0 = tm * 16, n0 = tn * 16;
  if (m0 >= M) return;                       // whole-wave uniform exit
  const int half = lane >> 4;
  const int lrow = lane & 15;
  const int rowA = m0 + lrow;
  const int rowB = n0 + lrow;
  const bool va = rowA < M;                  // m0 < M and n0 < N always hold,
  const bool vb = rowB < N;                  // so clamped rows are in bounds
  const float amask = va ? 1.f : 0.f;
  const float bmask = vb ? 1.f : 0.f;
  const float* __restrict__ pa = A + (long)(va ? rowA : m0) * K;
  const float* __restrict__ pb = B + (long)(vb ? rowB : n0) * K;

  v8f acc = {};
  const int Kfull = K & ~31;
  int k0 = 0;
  // ---- full 32-wide K chunks: plain contiguous loads (coalescible) ----
  for (; k0 < Kfull; k0 += 32) {
    v16h af, bf;
#pragma unroll
    for (int e = 0; e < 16; ++e) {
      // A 16x32 fragment (16-bit): k = (e/8)*16 + half*8 + e%8
      int ka = k0 + ((e >> 3) << 4) + (half << 3) + (e & 7);
      // B 32x16 fragment (16-bit): k = (e/8)*16 + half*8 + e%8
      int kb = k0 + ((e >> 3) << 4) + (half << 3) + (e & 7);
      af[e] = (_Float16)(pa[ka] * amask);
      bf[e] = (_Float16)(pb[kb] * bmask);
    }
    acc = __builtin_amdgcn_wmma_f32_16x16x32_f16(
        /*neg_a=*/false, af, /*neg_b=*/false, bf,
        /*c_mod=*/(short)0, acc, /*reuse_a=*/false, /*reuse_b=*/false);
  }
  // ---- ragged K tail: clamped indices + mask multiply, still branchless ----
  if (k0 < K) {
    v16h af, bf;
#pragma unroll
    for (int e = 0; e < 16; ++e) {
      int ka = k0 + ((e >> 3) << 4) + (half << 3) + (e & 7);
      int kb = k0 + ((e >> 3) << 4) + (half << 3) + (e & 7);
      int kac = ka < K ? ka : K - 1;
      int kbc = kb < K ? kb : K - 1;
      af[e] = (_Float16)(pa[kac] * (ka < K ? amask : 0.f));
      bf[e] = (_Float16)(pb[kbc] * (kb < K ? bmask : 0.f));
    }
    acc = __builtin_amdgcn_wmma_f32_16x16x32_f16(
        false, af, false, bf, (short)0, acc, false, false);
  }

  const int n = n0 + lrow;
  if (n < N) {
    float bv = bias ? bias[n] : 0.f;
#pragma unroll
    for (int r = 0; r < 8; ++r) {            // D layout: n=lane&15, m=r+8*half
      int m = m0 + r + half * 8;
      if (m < M) C[(long)m * N + n] = acc[r] + bv;
    }
  }
}

// --------------------------------------------------------------- GAT edge ops
// Pass A: alpha[e,h] = <att_h, leaky_relu(xl[src]+xr[dst]+We@ea)> ; segment max
__global__ void k_edge_alpha_max(const int* __restrict__ ei, const float* __restrict__ eattr,
                                 const float* __restrict__ xl, const float* __restrict__ xr,
                                 const float* __restrict__ We, const float* __restrict__ att,
                                 float* __restrict__ alpha, unsigned* __restrict__ amax,
                                 int E, int H, int C) {
  long t = (long)blockIdx.x * blockDim.x + threadIdx.x;
  if (t >= (long)E * H) return;
  int e = (int)(t / H), h = (int)(t % H);
  int src = ei[e], dst = ei[E + e];
  int hc = H * C;
  const float* xls = xl + (long)src * hc + h * C;
  const float* xrd = xr + (long)dst * hc + h * C;
  const float* at  = att + h * C;
  float ea[9];
#pragma unroll
  for (int j = 0; j < 9; ++j) ea[j] = eattr[(long)e * 9 + j];
  float s = 0.f;
  for (int c = 0; c < C; ++c) {
    const float* w = We + (long)(h * C + c) * 9;
    float ev = 0.f;
#pragma unroll
    for (int j = 0; j < 9; ++j) ev += ea[j] * w[j];
    float mv = xls[c] + xrd[c] + ev;
    mv = mv > 0.f ? mv : 0.2f * mv;          // leaky_relu(0.2)
    s += mv * at[c];
  }
  alpha[t] = s;
  atomicMax(&amax[(long)dst * H + h], f2mono(s));
}

// Pass B: ex = exp(alpha - amax[dst]); den[dst] += ex  (alpha overwritten by ex)
__global__ void k_edge_softmax_den(const int* __restrict__ ei, float* __restrict__ alpha,
                                   const unsigned* __restrict__ amax, float* __restrict__ den,
                                   int E, int H) {
  long t = (long)blockIdx.x * blockDim.x + threadIdx.x;
  if (t >= (long)E * H) return;
  int e = (int)(t / H), h = (int)(t % H);
  int dst = ei[E + e];
  float am = mono2f(amax[(long)dst * H + h]);
  float ex = __expf(alpha[t] - am);
  alpha[t] = ex;
  atomicAdd(&den[(long)dst * H + h], ex);
}

// Pass C: out[dst] += (ex/den[dst]) * xl[src]
__global__ void k_edge_aggregate(const int* __restrict__ ei, const float* __restrict__ alpha,
                                 const float* __restrict__ den, const float* __restrict__ xl,
                                 float* __restrict__ out, int E, int H, int C) {
  long t = (long)blockIdx.x * blockDim.x + threadIdx.x;
  if (t >= (long)E * H) return;
  int e = (int)(t / H), h = (int)(t % H);
  int src = ei[e], dst = ei[E + e];
  int hc = H * C;
  float a = alpha[t] / (den[(long)dst * H + h] + 1e-16f);
  const float* xs = xl + (long)src * hc + h * C;
  float* od = out + (long)dst * hc + h * C;
  for (int c = 0; c < C; ++c) atomicAdd(&od[c], xs[c] * a);
}

// -------------------------------------------------------------- GraphNorm
// Per-column sum & sum-of-squares via LDS (ds_add_f32) then one global atomic
__global__ void k_colstats(const float* __restrict__ x, float* __restrict__ sum,
                           float* __restrict__ sq, long total, int D) {
  extern __shared__ float sm[];
  float* s0 = sm;
  float* s1 = sm + D;
  for (int i = threadIdx.x; i < 2 * D; i += blockDim.x) sm[i] = 0.f;
  __syncthreads();
  for (long i = (long)blockIdx.x * blockDim.x + threadIdx.x; i < total;
       i += (long)gridDim.x * blockDim.x) {
    float v = x[i];
    int d = (int)(i % D);
    atomicAdd(&s0[d], v);
    atomicAdd(&s1[d], v * v);
  }
  __syncthreads();
  for (int d = threadIdx.x; d < D; d += blockDim.x) {
    atomicAdd(&sum[d], s0[d]);
    atomicAdd(&sq[d], s1[d]);
  }
}

// y = w*(x - ms*mean)*rsqrt(var+1e-5)+b,  var = E[x^2] - 2*ms*mu^2 + ms^2*mu^2
__global__ void k_gn_apply(float* __restrict__ x, const float* __restrict__ sum,
                           const float* __restrict__ sq, const float* __restrict__ w,
                           const float* __restrict__ b, const float* __restrict__ ms,
                           long total, int Nrows, int D) {
  long i = (long)blockIdx.x * blockDim.x + threadIdx.x;
  if (i >= total) return;
  int d = (int)(i % D);
  float invn = 1.f / (float)Nrows;
  float mu  = sum[d] * invn;
  float ex2 = sq[d] * invn;
  float m = ms[d];
  float var = ex2 - 2.f * m * mu * mu + m * m * mu * mu;
  float y = x[i] - m * mu;
  x[i] = w[d] * y * rsqrtf(var + 1e-5f) + b[d];
}

// ------------------------------------------------------------- mean pool
__global__ void k_pool_accum(const float* __restrict__ x, const int* __restrict__ batch,
                             float* __restrict__ pool, float* __restrict__ cnt,
                             int Nrows, int D) {
  int n = blockIdx.x * blockDim.x + threadIdx.x;
  if (n >= Nrows) return;
  int b = batch[n];
  const float* xr = x + (long)n * D;
  float* pr = pool + (long)b * D;
  for (int j = 0; j < D; ++j) atomicAdd(&pr[j], xr[j]);
  atomicAdd(&cnt[b], 1.f);
}
__global__ void k_pool_final(float* __restrict__ pool, const float* __restrict__ cnt,
                             int total, int D) {
  int i = blockIdx.x * blockDim.x + threadIdx.x;
  if (i >= total) return;
  pool[i] /= fmaxf(cnt[i / D], 1.f);
}

// ------------------------------------------------------- BatchNorm1d (train)
__global__ void k_bn_cols(const float* __restrict__ z, float* __restrict__ z2,
                          const float* __restrict__ w, const float* __restrict__ b,
                          int M, int Ncol) {
  int c = blockIdx.x * blockDim.x + threadIdx.x;
  if (c >= Ncol) return;
  float s = 0.f, sq = 0.f;
  for (int r = 0; r < M; ++r) { float v = z[(long)r * Ncol + c]; s += v; sq += v * v; }
  float mu = s / (float)M;
  float var = sq / (float)M - mu * mu;
  float inv = rsqrtf(var + 1e-5f);
  float ww = w[c], bb = b[c];
  for (int r = 0; r < M; ++r)
    z2[(long)r * Ncol + c] = ww * (z[(long)r * Ncol + c] - mu) * inv + bb;
}

__global__ void k_sigmoid(const float* __restrict__ z, float* __restrict__ out, int n) {
  int i = blockIdx.x * blockDim.x + threadIdx.x;
  if (i >= n) return;
  out[i] = 1.f / (1.f + __expf(-z[i]));
}

// ============================================================ host orchestration
static inline int nblk(long n, int b) { long k = (n + b - 1) / b; return (int)(k < 1 ? 1 : k); }
static inline int capblk(long n, int b, int cap) { int k = nblk(n, b); return k > cap ? cap : k; }

static void gemm_nt(const float* A, const float* B, const float* bias, float* C,
                    int M, int N, int K, hipStream_t s) {
  int tm = (M + 15) / 16, tn = (N + 15) / 16;
  long tiles = (long)tm * tn;
  int waves_per_block = 4;
  int blocks = (int)((tiles + waves_per_block - 1) / waves_per_block);
  k_wmma_gemm_nt<<<blocks, waves_per_block * 32, 0, s>>>(A, B, bias, C, M, N, K, tn);
}

static void graphnorm(float* x, int Nrows, int D, const float* w, const float* b,
                      const float* ms, float* sum, float* sq, hipStream_t s) {
  k_fill_f32<<<1, 256, 0, s>>>(sum, 0.f, D);
  k_fill_f32<<<1, 256, 0, s>>>(sq, 0.f, D);
  long total = (long)Nrows * D;
  k_colstats<<<capblk(total, 256, 2048), 256, 2 * D * sizeof(float), s>>>(x, sum, sq, total, D);
  k_gn_apply<<<nblk(total, 256), 256, 0, s>>>(x, sum, sq, w, b, ms, total, Nrows, D);
}

struct GatP { const float *Wl, *bl, *Wr, *br, *We, *att, *bias; };

static void run_gat(const float* x, const int* ei, const float* eattr,
                    int Nn, int E, int inDim, int H, int C, const GatP& p,
                    float* xl, float* xr, float* alpha, unsigned* amax, float* den,
                    float* out, hipStream_t s) {
  int hc = H * C;
  // dense transforms on WMMA
  gemm_nt(x, p.Wl, p.bl, xl, Nn, hc, inDim, s);
  gemm_nt(x, p.Wr, p.br, xr, Nn, hc, inDim, s);
  // init accumulators
  k_fill_u32<<<capblk((long)Nn * H, 256, 2048), 256, 0, s>>>(amax, 0u, (long)Nn * H);
  k_fill_f32<<<capblk((long)Nn * H, 256, 2048), 256, 0, s>>>(den, 0.f, (long)Nn * H);
  k_bias_init<<<capblk((long)Nn * hc, 256, 4096), 256, 0, s>>>(out, p.bias, (long)Nn * hc, hc);
  // scatter-softmax attention
  long eh = (long)E * H;
  k_edge_alpha_max<<<nblk(eh, 256), 256, 0, s>>>(ei, eattr, xl, xr, p.We, p.att,
                                                 alpha, amax, E, H, C);
  k_edge_softmax_den<<<nblk(eh, 256), 256, 0, s>>>(ei, alpha, amax, den, E, H);
  k_edge_aggregate<<<nblk(eh, 256), 256, 0, s>>>(ei, alpha, den, xl, out, E, H, C);
}

extern "C" void kernel_launch(void* const* d_in, const int* in_sizes, int n_in,
                              void* d_out, int out_size, void* d_ws, size_t ws_size,
                              hipStream_t stream) {
  (void)in_sizes; (void)n_in; (void)out_size; (void)ws_size;
  const float* x_s   = (const float*)d_in[0];
  const int*   ei_s  = (const int*)d_in[1];
  const float* ea_s  = (const float*)d_in[2];
  // d_in[3..5]: x_t / edge_index_t / edge_attr_t -> dead code (output only uses xs)
  const int*   bat_s = (const int*)d_in[6];
  // d_in[7]: xt_batch -> dead

  // params flattened in dict insertion order starting at index 8
  int ix = 8;
  GatP s1 { (const float*)d_in[ix+0], (const float*)d_in[ix+1], (const float*)d_in[ix+2],
            (const float*)d_in[ix+3], (const float*)d_in[ix+4], (const float*)d_in[ix+5],
            (const float*)d_in[ix+6] };
  ix += 7;
  const float* gn1w = (const float*)d_in[ix+0];
  const float* gn1b = (const float*)d_in[ix+1];
  const float* gn1m = (const float*)d_in[ix+2];
  ix += 3;
  GatP s2 { (const float*)d_in[ix+0], (const float*)d_in[ix+1], (const float*)d_in[ix+2],
            (const float*)d_in[ix+3], (const float*)d_in[ix+4], (const float*)d_in[ix+5],
            (const float*)d_in[ix+6] };
  ix += 7;
  const float* gn2w = (const float*)d_in[ix+0];
  const float* gn2b = (const float*)d_in[ix+1];
  const float* gn2m = (const float*)d_in[ix+2];
  ix += 3;
  ix += 7 + 3 + 7 + 3;   // skip t1 / gn3 / t2 / gn4 (dead)
  const float* L1w  = (const float*)d_in[ix+0];
  const float* L1b  = (const float*)d_in[ix+1];
  const float* bn_w = (const float*)d_in[ix+2];
  const float* bn_b = (const float*)d_in[ix+3];
  const float* L2w  = (const float*)d_in[ix+4];
  const float* L2b  = (const float*)d_in[ix+5];

  // ---------------- workspace layout (float offsets) ----------------
  float* ws = (float*)d_ws;
  const long O_XL     = 0;                         // 30000*256
  const long O_XR     = O_XL     + 7680000;
  const long O_H1     = O_XR     + 7680000;
  const long O_ALPHA  = O_H1     + 7680000;        // 480000*8
  const long O_AMAX   = O_ALPHA  + 3840000;        // 30000*8 (u32)
  const long O_DEN    = O_AMAX   + 240000;
  const long O_XL2    = O_DEN    + 240000;         // 30000*16
  const long O_XR2    = O_XL2    + 480000;
  const long O_H2     = O_XR2    + 480000;
  const long O_ALPHA2 = O_H2     + 480000;         // 480000*1
  const long O_AMAX2  = O_ALPHA2 + 480000;         // 30000 (u32)
  const long O_DEN2   = O_AMAX2  + 30000;
  const long O_SUM    = O_DEN2   + 30000;          // 256
  const long O_SQ     = O_SUM    + 256;
  const long O_POOL   = O_SQ     + 256;            // 64*16
  const long O_CNT    = O_POOL   + 1024;           // 64
  const long O_Z      = O_CNT    + 64;             // 64*658
  const long O_Z2     = O_Z      + 42112;

  float* zout = (float*)d_out;                     // [64,1317], then sigmoid copy

  // ---------------- S tower, layer 1: GATv2(64 -> 8 heads x 32) ----------------
  run_gat(x_s, ei_s, ea_s, NS, ES, 64, 8, 32, s1,
          ws + O_XL, ws + O_XR, ws + O_ALPHA, (unsigned*)(ws + O_AMAX), ws + O_DEN,
          ws + O_H1, stream);
  graphnorm(ws + O_H1, NS, 256, gn1w, gn1b, gn1m, ws + O_SUM, ws + O_SQ, stream);

  // ---------------- S tower, layer 2: GATv2(256 -> 1 head x 16) ----------------
  run_gat(ws + O_H1, ei_s, ea_s, NS, ES, 256, 1, 16, s2,
          ws + O_XL2, ws + O_XR2, ws + O_ALPHA2, (unsigned*)(ws + O_AMAX2), ws + O_DEN2,
          ws + O_H2, stream);
  graphnorm(ws + O_H2, NS, 16, gn2w, gn2b, gn2m, ws + O_SUM, ws + O_SQ, stream);

  // ---------------- mean pool over batch -> xs [64,16] ----------------
  k_fill_f32<<<nblk(1024 + 64, 256), 256, 0, stream>>>(ws + O_POOL, 0.f, 1024 + 64);
  k_pool_accum<<<nblk(NS, 256), 256, 0, stream>>>(ws + O_H2, bat_s, ws + O_POOL,
                                                  ws + O_CNT, NS, 16);
  k_pool_final<<<nblk(1024, 256), 256, 0, stream>>>(ws + O_POOL, ws + O_CNT, 1024, 16);

  // ---------------- head: Linear(16->658) + BN + Linear(658->1317) -------------
  gemm_nt(ws + O_POOL, L1w, L1b, ws + O_Z, NB, 658, 16, stream);
  k_bn_cols<<<nblk(658, 256), 256, 0, stream>>>(ws + O_Z, ws + O_Z2, bn_w, bn_b, NB, 658);
  gemm_nt(ws + O_Z2, L2w, L2b, zout, NB, 1317, 658, stream);

  const int ZN = NB * 1317;
  k_sigmoid<<<nblk(ZN, 256), 256, 0, stream>>>(zout, zout + ZN, ZN);
}
